// Attention_91293824843977
// MI455X (gfx1250) — matrix operand order repaired and checked by reading, not compiled, not run on hardware
//
#include <hip/hip_runtime.h>

#define EMBED 384
#define NHEAD 6
#define HDIM  64
#define BB    8
#define TT    64
#define NN    196
#define NTOK  (BB*TT*NN)      // 100352
#define QKV_O (3*EMBED)       // 1152
#define ATT_SCALE 0.125f
#define NPAD  224

typedef __attribute__((ext_vector_type(16))) __bf16         v16bf;
typedef __attribute__((ext_vector_type(8)))  float          v8f;
typedef __attribute__((ext_vector_type(8)))  unsigned short v8u16;

union Frag16 { v16bf bf; v8u16 u[2]; };

__device__ __forceinline__ unsigned short f2bf(float f) {
  unsigned int u = __builtin_bit_cast(unsigned int, f);
  u += 0x7fffu + ((u >> 16) & 1u);          // round-to-nearest-even
  return (unsigned short)(u >> 16);
}

__device__ __forceinline__ v8f v8f_zero() {
  v8f r;
#pragma unroll
  for (int i = 0; i < 8; ++i) r[i] = 0.f;
  return r;
}

// ---- DPP row-rotate reductions (16-lane rows, matches WMMA C row groups) ----
template <int CTRL>
__device__ __forceinline__ float dpp_rotf(float x) {
  int s = __builtin_bit_cast(int, x);
  int r = __builtin_amdgcn_update_dpp(s, s, CTRL, 0xF, 0xF, false);
  return __builtin_bit_cast(float, r);
}
__device__ __forceinline__ float rowmax16(float x) {
  x = fmaxf(x, dpp_rotf<0x121>(x));   // row_ror:1
  x = fmaxf(x, dpp_rotf<0x122>(x));   // row_ror:2
  x = fmaxf(x, dpp_rotf<0x124>(x));   // row_ror:4
  x = fmaxf(x, dpp_rotf<0x128>(x));   // row_ror:8
  return x;
}
__device__ __forceinline__ float rowsum16(float x) {
  x += dpp_rotf<0x121>(x);
  x += dpp_rotf<0x122>(x);
  x += dpp_rotf<0x124>(x);
  x += dpp_rotf<0x128>(x);
  return x;
}

// ---- CDNA5 async global -> LDS (ASYNCcnt-tracked), via inline asm ----
__device__ __forceinline__ void async_load_b128(const void* gptr, void* lptr) {
  unsigned lds = (unsigned)(size_t)lptr;                 // low 32 bits = LDS offset
  unsigned long long ga = (unsigned long long)(size_t)gptr;
  asm volatile("global_load_async_to_lds_b128 %0, %1, off"
               :: "v"(lds), "v"(ga) : "memory");
}
template <int N>
__device__ __forceinline__ void wait_async() {
  asm volatile("s_wait_asynccnt %0" :: "i"(N) : "memory");
}

// ---------------------------------------------------------------------------
// fp32 -> bf16 conversion (weights only; x is converted in-GEMM)
// ---------------------------------------------------------------------------
__global__ void f2bf_kernel(const float* __restrict__ in,
                            unsigned short* __restrict__ out, int n) {
  int i = blockIdx.x * blockDim.x + threadIdx.x;
  if (i < n) out[i] = f2bf(in[i]);
}

// ---------------------------------------------------------------------------
// GEMM: Out[m,n] = sum_k A[m,k] * Bw[n,k] + bias[n]
//   A_F32:  fp32 A, register-prefetch pipeline + in-register bf16 convert
//   !A_F32: bf16 A, double-buffered async global->LDS pipeline (ASYNCcnt)
//   Bw: [N,K] bf16, row-major (K contiguous) - L2 resident, read per-lane
// Block: 128x128 tile, 256 threads = 8 waves, wave tile 32(M) x 64(N).
// ---------------------------------------------------------------------------
template <bool A_F32, bool OUT_BF16>
__global__ __launch_bounds__(256) void gemm_bf16(
    const void* __restrict__ Aptr, const unsigned short* __restrict__ Bw,
    const float* __restrict__ bias, void* __restrict__ Out,
    int M, int N, int K) {
  __shared__ __align__(16) unsigned short As[2][128 * 40];  // 32 halfs + 8 pad

  const int tid  = threadIdx.x;
  const int lane = tid & 31;
  const int wave = tid >> 5;
  const int wm   = (wave & 3) * 32;   // 0..96
  const int wn   = (wave >> 2) * 64;  // 0,64
  const int mBase = blockIdx.x * 128;
  const int nBase = blockIdx.y * 128;
  const int KSTEPS = K >> 5;          // K/32

  v8f acc[2][4];
#pragma unroll
  for (int i = 0; i < 2; ++i)
#pragma unroll
    for (int j = 0; j < 4; ++j) acc[i][j] = v8f_zero();

  const int rRow = tid >> 1;   // 0..127
  const int rSeg = tid & 1;    // which 16-half chunk

  const float* aRowF = (const float*)Aptr + (size_t)(mBase + rRow) * K + rSeg * 16;
  const unsigned short* aRowH =
      (const unsigned short*)Aptr + (size_t)(mBase + rRow) * K + rSeg * 16;
  unsigned short* dst0 = &As[0][rRow * 40 + rSeg * 16];

  float4 pf[4];
  if (A_F32) {
    const float4* p = (const float4*)aRowF;
#pragma unroll
    for (int i = 0; i < 4; ++i) pf[i] = p[i];
  } else {
    // prologue: async-stage tile 0 into buffer 0
    async_load_b128(aRowH,     dst0);
    async_load_b128(aRowH + 8, dst0 + 8);
  }

  for (int step = 0; step < KSTEPS; ++step) {
    const int kk = step * 32;
    const unsigned short* curBuf = As[step & 1];

    if (A_F32) {
      // convert prefetched regs -> LDS (single buffer, barrier-protected)
      const float* pfs = (const float*)pf;
      unsigned short tmp[16];
#pragma unroll
      for (int i = 0; i < 16; ++i) tmp[i] = f2bf(pfs[i]);
      unsigned short* d = &As[0][rRow * 40 + rSeg * 16];
      *(v8u16*)&d[0] = *(v8u16*)&tmp[0];
      *(v8u16*)&d[8] = *(v8u16*)&tmp[8];
      curBuf = As[0];
      __syncthreads();
      if (step + 1 < KSTEPS) {   // overlap next tile's global loads with WMMA
        const float4* p = (const float4*)(aRowF + kk + 32);
#pragma unroll
        for (int i = 0; i < 4; ++i) pf[i] = p[i];
      }
    } else {
      __syncthreads();           // all reads of the buffer we are about to fill done
      if (step + 1 < KSTEPS) {
        unsigned short* d = &As[(step + 1) & 1][rRow * 40 + rSeg * 16];
        const unsigned short* src = aRowH + kk + 32;
        async_load_b128(src,     d);
        async_load_b128(src + 8, d + 8);
        wait_async<2>();         // tile `step` landed (only next tile outstanding)
      } else {
        wait_async<0>();
      }
      __syncthreads();           // every wave's portion of tile `step` visible
    }

    // A fragments: lane m = lane%16; elems 0..7 = K (lane/16)*8.., 8..15 = K 16+..
    Frag16 a[2];
#pragma unroll
    for (int mt = 0; mt < 2; ++mt) {
      int base = (wm + mt * 16 + (lane & 15)) * 40 + (lane >> 4) * 8;
      a[mt].u[0] = *(const v8u16*)&curBuf[base];
      a[mt].u[1] = *(const v8u16*)&curBuf[base + 16];
    }
    // B fragments direct from global (L2-hot weights): lane col = lane%16,
    // elems 0..7 = K (lane/16)*8.., 8..15 = K 16+(lane/16)*8..
#pragma unroll
    for (int nt = 0; nt < 4; ++nt) {
      int col = nBase + wn + nt * 16 + (lane & 15);
      const unsigned short* bp = Bw + (size_t)col * K + kk + (lane >> 4) * 8;
      Frag16 bf;
      bf.u[0] = *(const v8u16*)&bp[0];
      bf.u[1] = *(const v8u16*)&bp[16];
#pragma unroll
      for (int mt = 0; mt < 2; ++mt) {
        acc[mt][nt] = __builtin_amdgcn_wmma_f32_16x16x32_bf16(
            false, a[mt].bf, false, bf.bf, (short)0, acc[mt][nt], false, false);
      }
    }
    if (A_F32) __syncthreads();  // protect single buffer before next overwrite
  }

  // ---- epilogue: C layout row = r + 8*(lane/16), col = lane%16 ----
#pragma unroll
  for (int mt = 0; mt < 2; ++mt)
#pragma unroll
    for (int nt = 0; nt < 4; ++nt) {
      int colN = nBase + wn + nt * 16 + (lane & 15);
      float bv = bias[colN];
#pragma unroll
      for (int rr = 0; rr < 8; ++rr) {
        int rowM = mBase + wm + mt * 16 + rr + 8 * (lane >> 4);
        float v = acc[mt][nt][rr] + bv;
        if (OUT_BF16)
          ((unsigned short*)Out)[(size_t)rowM * N + colN] = f2bf(v);
        else
          ((float*)Out)[(size_t)rowM * N + colN] = v;
      }
    }
}

// ---------------------------------------------------------------------------
// Flash attention over one (b, h, t) block: q,k,v are [196,64] bf16 slices of
// the qkv buffer. grid = (B*H*T, 2); block = 224 threads (7 waves), each wave
// owns one 16-row query tile (13 tiles total across grid.y).
// ctx output uses the reference's permuted channel layout: c = d*NHEAD + h.
// ---------------------------------------------------------------------------
__global__ __launch_bounds__(224) void attn_flash(
    const unsigned short* __restrict__ qkv, unsigned short* __restrict__ ctx) {
  __shared__ __align__(16) unsigned short kbuf[NPAD * 64];   // K row-major, zero padded
  __shared__ __align__(16) unsigned short vtb[64 * NPAD];    // V transposed [d][key]
  __shared__ __align__(16) unsigned short ps[7][16 * 32];    // per-wave P relayout

  const int tid = threadIdx.x;
  const int bht = blockIdx.x;
  const int t   = bht % TT;
  const int hh  = (bht / TT) % NHEAD;
  const int b   = bht / (TT * NHEAD);
  const size_t tokBase = ((size_t)b * TT + t) * NN;

  // ---- K rows: async global->LDS; V rows: load + transpose-scatter ----
  for (int i = tid; i < NPAD * 8; i += 224) {
    int n = i >> 3, seg = i & 7;
    v8u16 vv;
    if (n < NN) {
      const unsigned short* kp =
          qkv + (tokBase + n) * QKV_O + EMBED + hh * HDIM + seg * 8;
      async_load_b128(kp, &kbuf[n * 64 + seg * 8]);
      vv = *(const v8u16*)(kp + EMBED);
    } else {
      v8u16 kz;
#pragma unroll
      for (int j = 0; j < 8; ++j) { kz[j] = 0; vv[j] = 0; }
      *(v8u16*)&kbuf[n * 64 + seg * 8] = kz;
    }
#pragma unroll
    for (int j = 0; j < 8; ++j) vtb[(seg * 8 + j) * NPAD + n] = vv[j];
  }
  wait_async<0>();
  __syncthreads();

  const int wid  = tid >> 5;
  const int lane = tid & 31;
  const int qtile = blockIdx.y * 7 + wid;
  if (qtile >= 13) return;   // no barriers past this point
  const int qb = qtile * 16;

  // ---- load Q A-fragments (two K=32 slices of hd=64), rows clamped ----
  int qm = qb + (lane & 15);
  if (qm > NN - 1) qm = NN - 1;
  const unsigned short* qp = qkv + (tokBase + qm) * QKV_O + hh * HDIM;
  Frag16 qa[2];
#pragma unroll
  for (int kt = 0; kt < 2; ++kt) {
    qa[kt].u[0] = *(const v8u16*)(qp + kt * 32 + (lane >> 4) * 8);
    qa[kt].u[1] = *(const v8u16*)(qp + kt * 32 + 16 + (lane >> 4) * 8);
  }

  float mrow[8], lrow[8];
  v8f oacc[4];
#pragma unroll
  for (int rr = 0; rr < 8; ++rr) { mrow[rr] = -1e30f; lrow[rr] = 0.f; }
#pragma unroll
  for (int dt = 0; dt < 4; ++dt) oacc[dt] = v8f_zero();

  for (int kp = 0; kp < 7; ++kp) {           // 7 x 32 keys (padded to 224)
    const int keyBase = kp * 32;
    // ---- S = q . k^T (two 16-col tiles, chained over hd) ----
    v8f s[2];
    s[0] = v8f_zero(); s[1] = v8f_zero();
#pragma unroll
    for (int ct = 0; ct < 2; ++ct) {
#pragma unroll
      for (int kt = 0; kt < 2; ++kt) {
        const unsigned short* kpp =
            &kbuf[(keyBase + ct * 16 + (lane & 15)) * 64 + kt * 32 + (lane >> 4) * 8];
        Frag16 kb;
        kb.u[0] = *(const v8u16*)&kpp[0];
        kb.u[1] = *(const v8u16*)&kpp[16];
        s[ct] = __builtin_amdgcn_wmma_f32_16x16x32_bf16(
            false, qa[kt].bf, false, kb.bf, (short)0, s[ct], false, false);
      }
    }
    // ---- scale + key mask ----
#pragma unroll
    for (int ct = 0; ct < 2; ++ct) {
      bool valid = (keyBase + ct * 16 + (lane & 15)) < NN;
#pragma unroll
      for (int rr = 0; rr < 8; ++rr)
        s[ct][rr] = valid ? s[ct][rr] * ATT_SCALE : -1e30f;
    }
    // ---- online softmax: DPP row reductions, zero LDS traffic ----
#pragma unroll
    for (int rr = 0; rr < 8; ++rr) {
      float x = rowmax16(fmaxf(s[0][rr], s[1][rr]));
      float mnew  = fmaxf(mrow[rr], x);
      float alpha = __expf(mrow[rr] - mnew);
      mrow[rr] = mnew;
      float p0 = __expf(s[0][rr] - mnew);
      float p1 = __expf(s[1][rr] - mnew);
      s[0][rr] = p0; s[1][rr] = p1;
      lrow[rr] = lrow[rr] * alpha + rowsum16(p0 + p1);
#pragma unroll
      for (int dt = 0; dt < 4; ++dt) oacc[dt][rr] *= alpha;
    }
    // ---- relayout P (C layout -> A layout) through per-wave LDS ----
#pragma unroll
    for (int ct = 0; ct < 2; ++ct)
#pragma unroll
      for (int rr = 0; rr < 8; ++rr)
        ps[wid][(rr + 8 * (lane >> 4)) * 32 + ct * 16 + (lane & 15)] = f2bf(s[ct][rr]);
    Frag16 pa;
    {
      int base = (lane & 15) * 32 + (lane >> 4) * 8;
      pa.u[0] = *(const v8u16*)&ps[wid][base];
      pa.u[1] = *(const v8u16*)&ps[wid][base + 16];
    }
    // ---- O += P . V ----
#pragma unroll
    for (int dt = 0; dt < 4; ++dt) {
      const unsigned short* vp =
          &vtb[(dt * 16 + (lane & 15)) * NPAD + keyBase + (lane >> 4) * 8];
      Frag16 vb;
      vb.u[0] = *(const v8u16*)&vp[0];
      vb.u[1] = *(const v8u16*)&vp[16];
      oacc[dt] = __builtin_amdgcn_wmma_f32_16x16x32_bf16(
          false, pa.bf, false, vb.bf, (short)0, oacc[dt], false, false);
    }
  }

  // ---- finalize + store ctx bf16 with permuted channels c = d*NHEAD + h ----
#pragma unroll
  for (int rr = 0; rr < 8; ++rr) {
    int n = qb + rr + 8 * (lane >> 4);
    if (n >= NN) continue;
    float inv = 1.0f / lrow[rr];
#pragma unroll
    for (int dt = 0; dt < 4; ++dt) {
      int d = dt * 16 + (lane & 15);
      ctx[(tokBase + n) * EMBED + d * NHEAD + hh] = f2bf(oacc[dt][rr] * inv);
    }
  }
}

// ---------------------------------------------------------------------------
extern "C" void kernel_launch(void* const* d_in, const int* in_sizes, int n_in,
                              void* d_out, int out_size, void* d_ws, size_t ws_size,
                              hipStream_t stream) {
  const float* x      = (const float*)d_in[0];
  const float* W_qkv  = (const float*)d_in[1];
  const float* b_qkv  = (const float*)d_in[2];
  const float* W_proj = (const float*)d_in[3];
  const float* b_proj = (const float*)d_in[4];

  char* ws = (char*)d_ws;
  unsigned short* Wq   = (unsigned short*)ws;                         // 1152*384 bf16
  unsigned short* Wp   = (unsigned short*)(ws + (1 << 20));           // 384*384 bf16
  unsigned short* qkvb = (unsigned short*)(ws + (2 << 20));           // NTOK*1152 bf16
  unsigned short* ctx  =
      (unsigned short*)(ws + (2 << 20) + (size_t)NTOK * QKV_O * 2);   // NTOK*384 bf16

  f2bf_kernel<<<(QKV_O * EMBED + 255) / 256, 256, 0, stream>>>(W_qkv, Wq, QKV_O * EMBED);
  f2bf_kernel<<<(EMBED * EMBED + 255) / 256, 256, 0, stream>>>(W_proj, Wp, EMBED * EMBED);

  // QKV projection: x fp32 converted in-kernel, bf16 WMMA, bf16 output
  gemm_bf16<true, true><<<dim3(NTOK / 128, QKV_O / 128), 256, 0, stream>>>(
      (const void*)x, Wq, b_qkv, (void*)qkvb, NTOK, QKV_O, EMBED);

  // Flash attention per (b,h,t); two q-halves per block row
  attn_flash<<<dim3(BB * TT * NHEAD, 2), 224, 0, stream>>>(qkvb, ctx);

  // Output projection: bf16 A, fp32 output
  gemm_bf16<false, false><<<dim3(NTOK / 128, EMBED / 128), 256, 0, stream>>>(
      (const void*)ctx, Wp, b_proj, d_out, NTOK, EMBED, EMBED);
}
